// AttentionWithKVCache_3075196584669
// MI455X (gfx1250) — matrix operand order repaired and checked by reading, not compiled, not run on hardware
//
#include <hip/hip_runtime.h>
#include <hip/hip_bf16.h>

// ---------------------------------------------------------------------------
// Problem constants (from reference)
// ---------------------------------------------------------------------------
#define DIMC   1024
#define NH     16
#define NKV    2
#define HD     64
#define REP    8       // NH / NKV
#define WIN    256
#define HALFW  128
#define BATCH  2
#define SEQ    2048
#define ROWS   (BATCH * SEQ)          // 4096
#define SPAD   2336                   // 2048 + 256 pad + 32 slack (tile-17 overreach)

typedef __attribute__((ext_vector_type(8)))  float        v8f;
typedef __attribute__((ext_vector_type(8)))  __bf16       v8bf;
typedef __attribute__((ext_vector_type(16))) __bf16       v16bf;
typedef __attribute__((ext_vector_type(4)))  unsigned int u32x4;
typedef __attribute__((ext_vector_type(8)))  unsigned int u32x8;

union BF16x16 { v16bf v; v8bf h[2]; };

#define WMMA_BF16(A, B, C) \
  __builtin_amdgcn_wmma_f32_16x16x32_bf16(false, (A), false, (B), (short)0, (C), false, false)

// 16-lane (half-wave) butterfly reductions: lanes 0-15 / 16-31 stay disjoint.
__device__ __forceinline__ float rowmax16(float v) {
  v = fmaxf(v, __shfl_xor(v, 1, 32));
  v = fmaxf(v, __shfl_xor(v, 2, 32));
  v = fmaxf(v, __shfl_xor(v, 4, 32));
  v = fmaxf(v, __shfl_xor(v, 8, 32));
  return v;
}
__device__ __forceinline__ float rowsum16(float v) {
  v += __shfl_xor(v, 1, 32);
  v += __shfl_xor(v, 2, 32);
  v += __shfl_xor(v, 4, 32);
  v += __shfl_xor(v, 8, 32);
  return v;
}

// ---------------------------------------------------------------------------
// Tensor Data Mover: 2-D tile DMA global -> LDS (bf16, `rows` x 64 elements,
// contiguous rows).  Descriptor per CDNA5 ISA ch.8 (D# groups 0/1), issued via
// inline asm so it is portable across the two toolchains' builtin arities.
// Tile dims == tensor dims -> no OOB handling needed.  workgroup_mask = 0.
// ---------------------------------------------------------------------------
__device__ __forceinline__ void tdm_load_rows64_bf16(unsigned lds_off,
                                                     const void* gptr,
                                                     unsigned rows) {
  unsigned long long ga = (unsigned long long)(size_t)gptr;
  u32x4 g0;
  g0[0] = 1u;                                      // count=1, is_restore=0
  g0[1] = lds_off;                                 // lds_addr [63:32]
  g0[2] = (unsigned)ga;                            // global_addr [95:64]
  g0[3] = (unsigned)(ga >> 32) | 0x80000000u;      // addr[56:32] | type=2 @127:126
  u32x8 g1;
  g1[0] = 0x10000u;                                // wg_mask=0, data_size=1 (2B)
  g1[1] = 64u << 16;                               // tensor_dim0=64 (lo16 @63:48)
  g1[2] = (rows & 0xFFFFu) << 16;                  // dim0 hi16=0 | tensor_dim1 lo16
  g1[3] = 64u << 16;                               // dim1 hi16=0 | tile_dim0=64
  g1[4] = rows & 0xFFFFu;                          // tile_dim1 | tile_dim2=0
  g1[5] = 64u;                                     // tensor_dim0_stride lo32 = 64
  g1[6] = 0u;                                      // stride hi16 | dim1_stride lo16
  g1[7] = 0u;                                      // dim1_stride hi32
  asm volatile("tensor_load_to_lds %0, %1" :: "s"(g0), "s"(g1) : "memory");
}

// ---------------------------------------------------------------------------
// Prep kernels
// ---------------------------------------------------------------------------
__global__ void k_convert_x(const float* __restrict__ x, __bf16* __restrict__ xb, int n) {
  int t = (blockIdx.x * blockDim.x + threadIdx.x) * 4;
  if (t + 3 < n) {
    float4 f = *(const float4*)(x + t);
    xb[t + 0] = (__bf16)f.x; xb[t + 1] = (__bf16)f.y;
    xb[t + 2] = (__bf16)f.z; xb[t + 3] = (__bf16)f.w;
  }
}

// Wt[n][k] = W[k][n], fp32 -> bf16.  K = in-dim, N = out-dim.
__global__ void k_transpose_w(const float* __restrict__ W, __bf16* __restrict__ Wt,
                              int K, int N) {
  int t = blockIdx.x * blockDim.x + threadIdx.x;
  if (t < N * K) {
    int nn = t / K, kk = t - nn * K;
    Wt[t] = (__bf16)W[kk * N + nn];
  }
}

__global__ void k_concat_bias(const float* __restrict__ bq, const float* __restrict__ bk,
                              const float* __restrict__ bv, float* __restrict__ dst) {
  int t = blockIdx.x * blockDim.x + threadIdx.x;
  if (t < DIMC)                dst[t] = bq[t];
  else if (t < DIMC + 128)     dst[t] = bk[t - DIMC];
  else if (t < DIMC + 256)     dst[t] = bv[t - DIMC - 128];
}

__global__ void k_zero_u32(unsigned int* __restrict__ p, int n32) {
  int t = blockIdx.x * blockDim.x + threadIdx.x;
  if (t < n32) p[t] = 0u;
}

// ---------------------------------------------------------------------------
// Fused QKV projection: xb(4096x1024) @ Wt^T(1280x1024) + bias, routed to
// Qb (row-major bf16), Kpad ([b][kv][SPAD][64], zero-padded), Vt ([b][kv][64][SPAD]).
// One wave computes a 16(M) x 64(N) tile with 4 WMMA accumulators.
// ---------------------------------------------------------------------------
__global__ void k_gemm_qkv(const __bf16* __restrict__ xb, const __bf16* __restrict__ Wt,
                           const float* __restrict__ bias,
                           __bf16* __restrict__ Qb, __bf16* __restrict__ Kpad,
                           __bf16* __restrict__ Vt) {
  const int wid  = threadIdx.x >> 5;
  const int lane = threadIdx.x & 31;
  const int task = blockIdx.x * 8 + wid;        // 5120 tasks
  const int mt = task / 20, nt = task - mt * 20;
  const int half = lane >> 4, n = lane & 15;

  const __bf16* arow = xb + (size_t)(mt * 16 + n) * DIMC;
  v8f acc[4] = {};

  for (int k0 = 0; k0 < DIMC; k0 += 32) {
    __builtin_prefetch(arow + k0 + 256, 0, 3);
    BF16x16 a;
    a.h[0] = *(const v8bf*)(arow + k0 + half * 16);
    a.h[1] = *(const v8bf*)(arow + k0 + half * 16 + 8);
#pragma unroll
    for (int c = 0; c < 4; ++c) {
      const int col = nt * 64 + c * 16 + n;
      v16bf bfr = *(const v16bf*)(Wt + (size_t)col * DIMC + k0 + half * 16);
      acc[c] = WMMA_BF16(a.v, bfr, acc[c]);
    }
  }

#pragma unroll
  for (int c = 0; c < 4; ++c) {
    const int col = nt * 64 + c * 16 + n;
    const float bv = bias[col];
#pragma unroll
    for (int r = 0; r < 8; ++r) {
      const int row = mt * 16 + 8 * half + r;
      const float val = acc[c][r] + bv;
      if (col < DIMC) {
        Qb[(size_t)row * DIMC + col] = (__bf16)val;
      } else {
        const int cc = col - DIMC;
        const int which = cc >> 7;            // 0 = K, 1 = V
        const int kv = (cc >> 6) & 1, d = cc & 63;
        const int bi = row >> 11, s = row & (SEQ - 1);
        if (which == 0)
          Kpad[((size_t)(bi * NKV + kv) * SPAD + HALFW + s) * HD + d] = (__bf16)val;
        else
          Vt[((size_t)(bi * NKV + kv) * HD + d) * SPAD + HALFW + s] = (__bf16)val;
      }
    }
  }
}

// ---------------------------------------------------------------------------
// Banded attention, flash-style. One wave = 16-query tile of one head.
// Window for query s covers padded keys [s, s+255]; for a 16-row tile that is
// key tiles kt=0..16 (17 tiles), processed as 9 pairs (tile 17 fully masked,
// its loads hit zeroed slack rows).  K tiles (32x64 bf16, 4 KB) are brought
// into LDS by the Tensor Data Mover; B-fragments are then ds_load'ed.
// ---------------------------------------------------------------------------
__global__ void k_attn(const __bf16* __restrict__ Qb, const __bf16* __restrict__ Kpad,
                       const __bf16* __restrict__ Vt, __bf16* __restrict__ attn_out) {
  __shared__ __bf16 Klds[8][32][64];           // per-wave K tile (32 KB total)
  __shared__ __bf16 Pbuf[8][16][32];           // per-wave P transpose buffer (8 KB)

  const int wid  = threadIdx.x >> 5;
  const int lane = threadIdx.x & 31;
  const int task = blockIdx.x * 8 + wid;       // 4096 tasks
  const int stile = task & 127;
  const int rep   = (task >> 7) & 7;
  const int kv    = (task >> 10) & 1;
  const int bi    = task >> 11;
  const int head  = kv * REP + rep;
  const int half = lane >> 4, n = lane & 15;
  const int qb = stile * 16;

  // Q fragments (A layout), d in [0,32) and [32,64)
  const __bf16* qrow = Qb + (size_t)(bi * SEQ + qb + n) * DIMC + head * HD;
  BF16x16 q0, q1;
  q0.h[0] = *(const v8bf*)(qrow + half * 16);
  q0.h[1] = *(const v8bf*)(qrow + half * 16 + 8);
  q1.h[0] = *(const v8bf*)(qrow + 32 + half * 16);
  q1.h[1] = *(const v8bf*)(qrow + 40 + half * 16);

  const __bf16* Kb = Kpad + (size_t)(bi * NKV + kv) * SPAD * HD;
  const __bf16* Vb = Vt   + (size_t)(bi * NKV + kv) * HD * SPAD;
  const unsigned klds_off = (unsigned)(size_t)&Klds[wid][0][0];

  float m[8], l[8];
  v8f o[4] = {};
#pragma unroll
  for (int r = 0; r < 8; ++r) { m[r] = -1e30f; l[r] = 0.0f; }

  for (int t = 0; t < 9; ++t) {
    const int jp0 = qb + 32 * t;

    // DMA this pair's 32x64 K tile into LDS via the Tensor Data Mover.
    tdm_load_rows64_bf16(klds_off, Kb + (size_t)jp0 * HD, 32u);
    __builtin_amdgcn_s_wait_tensorcnt(0);

    v8f s[2] = {};
#pragma unroll
    for (int sub = 0; sub < 2; ++sub) {
      const __bf16* krow = &Klds[wid][sub * 16 + n][0];
      v16bf b0 = *(const v16bf*)(krow + half * 16);
      v16bf b1 = *(const v16bf*)(krow + 32 + half * 16);
      s[sub] = WMMA_BF16(q0.v, b0, s[sub]);
      s[sub] = WMMA_BF16(q1.v, b1, s[sub]);
      const int kt = 2 * t + sub;
      if (kt == 0) {                     // valid iff col >= row
#pragma unroll
        for (int r = 0; r < 8; ++r) { if (n < 8 * half + r) s[sub][r] = -1e30f; }
      } else if (kt == 16) {             // valid iff col < row
#pragma unroll
        for (int r = 0; r < 8; ++r) { if (n >= 8 * half + r) s[sub][r] = -1e30f; }
      } else if (kt == 17) {             // fully outside window
#pragma unroll
        for (int r = 0; r < 8; ++r) s[sub][r] = -1e30f;
      }
    }

    float alpha[8];
#pragma unroll
    for (int r = 0; r < 8; ++r) {
      float mx = rowmax16(fmaxf(s[0][r], s[1][r]));
      float mn = fmaxf(m[r], mx);
      alpha[r] = __expf(m[r] - mn);
      m[r] = mn;
      float p0 = __expf(s[0][r] - mn);
      float p1 = __expf(s[1][r] - mn);
      s[0][r] = p0; s[1][r] = p1;
      l[r] = l[r] * alpha[r] + rowsum16(p0 + p1);
    }

    __syncthreads();                     // WAR vs previous iteration's reads
#pragma unroll
    for (int r = 0; r < 8; ++r) {
      Pbuf[wid][8 * half + r][n]      = (__bf16)s[0][r];
      Pbuf[wid][8 * half + r][16 + n] = (__bf16)s[1][r];
    }
    __syncthreads();                     // RAW: P visible before A-frag reads

    BF16x16 pa;
    pa.h[0] = *(const v8bf*)&Pbuf[wid][n][half * 16];
    pa.h[1] = *(const v8bf*)&Pbuf[wid][n][half * 16 + 8];
#pragma unroll
    for (int c = 0; c < 4; ++c) {
      v16bf vb = *(const v16bf*)(Vb + (size_t)(c * 16 + n) * SPAD + jp0 + half * 16);
#pragma unroll
      for (int r = 0; r < 8; ++r) o[c][r] *= alpha[r];
      o[c] = WMMA_BF16(pa.v, vb, o[c]);
    }
  }

#pragma unroll
  for (int r = 0; r < 8; ++r) {
    const float inv = 1.0f / l[r];
    const size_t row = (size_t)(bi * SEQ + qb + 8 * half + r) * DIMC + head * HD;
#pragma unroll
    for (int c = 0; c < 4; ++c)
      attn_out[row + c * 16 + n] = (__bf16)(o[c][r] * inv);
  }
}

// ---------------------------------------------------------------------------
// Output projection: attn_out(4096x1024)bf16 @ Wot^T + bo -> fp32 out.
// ---------------------------------------------------------------------------
__global__ void k_gemm_out(const __bf16* __restrict__ A, const __bf16* __restrict__ Wt,
                           const float* __restrict__ bo, float* __restrict__ out) {
  const int wid  = threadIdx.x >> 5;
  const int lane = threadIdx.x & 31;
  const int task = blockIdx.x * 8 + wid;        // 4096 tasks
  const int mt = task >> 4, nt = task & 15;
  const int half = lane >> 4, n = lane & 15;

  const __bf16* arow = A + (size_t)(mt * 16 + n) * DIMC;
  v8f acc[4] = {};

  for (int k0 = 0; k0 < DIMC; k0 += 32) {
    __builtin_prefetch(arow + k0 + 256, 0, 3);
    BF16x16 a;
    a.h[0] = *(const v8bf*)(arow + k0 + half * 16);
    a.h[1] = *(const v8bf*)(arow + k0 + half * 16 + 8);
#pragma unroll
    for (int c = 0; c < 4; ++c) {
      const int col = nt * 64 + c * 16 + n;
      v16bf bfr = *(const v16bf*)(Wt + (size_t)col * DIMC + k0 + half * 16);
      acc[c] = WMMA_BF16(a.v, bfr, acc[c]);
    }
  }

#pragma unroll
  for (int c = 0; c < 4; ++c) {
    const int col = nt * 64 + c * 16 + n;
    const float bv = bo[col];
#pragma unroll
    for (int r = 0; r < 8; ++r) {
      const int row = mt * 16 + 8 * half + r;
      out[(size_t)row * DIMC + col] = acc[c][r] + bv;
    }
  }
}

// ---------------------------------------------------------------------------
// Host launcher
// ---------------------------------------------------------------------------
extern "C" void kernel_launch(void* const* d_in, const int* in_sizes, int n_in,
                              void* d_out, int out_size, void* d_ws, size_t ws_size,
                              hipStream_t stream) {
  const float* x  = (const float*)d_in[0];
  const float* Wq = (const float*)d_in[1];
  const float* bq = (const float*)d_in[2];
  const float* Wk = (const float*)d_in[3];
  const float* bk = (const float*)d_in[4];
  const float* Wv = (const float*)d_in[5];
  const float* bv = (const float*)d_in[6];
  const float* Wo = (const float*)d_in[7];
  const float* bo = (const float*)d_in[8];
  float* out = (float*)d_out;

  // Workspace carve-up (~30.8 MB total)
  char* w = (char*)d_ws;
  __bf16* xb     = (__bf16*)w;  w += (size_t)ROWS * DIMC * 2;          // 8 MB
  __bf16* Wqkvt  = (__bf16*)w;  w += (size_t)1280 * DIMC * 2;          // 2.5 MB
  float*  bqkv   = (float*)w;   w += (size_t)1280 * 4;
  __bf16* Wot    = (__bf16*)w;  w += (size_t)DIMC * DIMC * 2;          // 2 MB
  __bf16* Qb     = (__bf16*)w;  w += (size_t)ROWS * DIMC * 2;          // 8 MB
  __bf16* Kpad   = (__bf16*)w;  w += (size_t)BATCH * NKV * SPAD * HD * 2;
  __bf16* Vt     = (__bf16*)w;  w += (size_t)BATCH * NKV * HD * SPAD * 2;
  __bf16* attn_o = (__bf16*)w;  w += (size_t)ROWS * DIMC * 2;          // 8 MB

  // 1) Conversions / transposes
  k_convert_x<<<(ROWS * DIMC) / (256 * 4), 256, 0, stream>>>(x, xb, ROWS * DIMC);
  k_transpose_w<<<(DIMC * DIMC) / 256, 256, 0, stream>>>(Wq, Wqkvt, DIMC, DIMC);
  k_transpose_w<<<(128 * DIMC) / 256, 256, 0, stream>>>(Wk, Wqkvt + (size_t)DIMC * DIMC, DIMC, 128);
  k_transpose_w<<<(128 * DIMC) / 256, 256, 0, stream>>>(Wv, Wqkvt + (size_t)1152 * DIMC, DIMC, 128);
  k_transpose_w<<<(DIMC * DIMC) / 256, 256, 0, stream>>>(Wo, Wot, DIMC, DIMC);
  k_concat_bias<<<5, 256, 0, stream>>>(bq, bk, bv, bqkv);

  // 2) Zero padded K/V buffers (pad rows must be exact zeros -> score 0, v 0)
  const int kvbytes = BATCH * NKV * SPAD * HD * 2;
  k_zero_u32<<<(kvbytes / 4 + 255) / 256, 256, 0, stream>>>((unsigned int*)Kpad, kvbytes / 4);
  k_zero_u32<<<(kvbytes / 4 + 255) / 256, 256, 0, stream>>>((unsigned int*)Vt, kvbytes / 4);

  // 3) Fused QKV projection (5120 wave-tasks / 8 waves per block)
  k_gemm_qkv<<<640, 256, 0, stream>>>(xb, Wqkvt, bqkv, Qb, Kpad, Vt);

  // 4) Banded attention (4096 wave-tasks)
  k_attn<<<512, 256, 0, stream>>>(Qb, Kpad, Vt, attn_o);

  // 5) Output projection (4096 wave-tasks)
  k_gemm_out<<<512, 256, 0, stream>>>(attn_o, Wot, bo, out);
}
